// EvolvedAttention_29661044146582
// MI455X (gfx1250) — matrix operand order repaired and checked by reading, not compiled, not run on hardware
//
#include <hip/hip_runtime.h>
#include <hip/hip_bf16.h>
#include <math.h>

#define BATCH   2
#define SEQ     2048
#define DMODEL  1024
#define NHEADS  16
#define DHEAD   64
#define MROWS   (BATCH*SEQ)           // 4096
#define TOT     (BATCH*SEQ*DMODEL)    // 4194304
#define TOPK    (SEQ/4)               // 512
#define LPAD    8                     // LDS row padding (bf16 elems, keeps 16B align)

typedef __bf16 bf16_t;
typedef __attribute__((ext_vector_type(16))) __bf16 v16bf;
typedef __attribute__((ext_vector_type(8)))  __bf16 v8bf;
typedef __attribute__((ext_vector_type(8)))  float  v8f;

// Build a 16-element bf16 fragment from two contiguous 16-byte chunks.
__device__ __forceinline__ v16bf frag16(const bf16_t* p0, const bf16_t* p1) {
  v8bf lo = *reinterpret_cast<const v8bf*>(p0);
  v8bf hi = *reinterpret_cast<const v8bf*>(p1);
  return __builtin_shufflevector(lo, hi, 0,1,2,3,4,5,6,7,8,9,10,11,12,13,14,15);
}

// gfx1250 async global->LDS copy (ASYNCcnt-tracked). LDS address is the low
// 32 bits of the generic pointer (LDS aperture encoding).
__device__ __forceinline__ void async_copy_b128(void* lds_ptr, const void* gptr) {
#if defined(__gfx1250__)
  unsigned lds_addr = (unsigned)(size_t)lds_ptr;
  asm volatile("global_load_async_to_lds_b128 %0, %1, off"
               :: "v"(lds_addr), "v"(gptr) : "memory");
#else
  (void)lds_ptr; (void)gptr;
#endif
}
__device__ __forceinline__ void wait_async0() {
#if defined(__gfx1250__)
  asm volatile("s_wait_asynccnt 0x0" ::: "memory");
#endif
}

__device__ __forceinline__ float wave_sum(float v) {
#pragma unroll
  for (int m = 16; m >= 1; m >>= 1) v += __shfl_xor(v, m, 32);
  return v;
}
__device__ __forceinline__ float wave_max(float v) {
#pragma unroll
  for (int m = 16; m >= 1; m >>= 1) v = fmaxf(v, __shfl_xor(v, m, 32));
  return v;
}
__device__ __forceinline__ int wave_isum(int v) {
#pragma unroll
  for (int m = 16; m >= 1; m >>= 1) v += __shfl_xor(v, m, 32);
  return v;
}

// ---------------------------------------------------------------- convert
__global__ __launch_bounds__(256) void f32_to_bf16_kernel(
    const float* __restrict__ in, bf16_t* __restrict__ out, int n) {
  int i = blockIdx.x * 256 + threadIdx.x;
  if (i < n) out[i] = (bf16_t)in[i];
}

// ------------------------------------------------- bf16 WMMA GEMM C=A*W+b
// A: [M,K] bf16, W: [K,N] fp32 (converted on load), C: [M,N] fp32.
// Block tile 64x64, k-step 32, 8 waves: wave w -> m-tile (w&3), n-tiles 2*(w>>2)(+1).
// A tile staged with gfx1250 async global->LDS; W tile converted + transposed.
__global__ __launch_bounds__(256) void gemm_bf16_kernel(
    const bf16_t* __restrict__ A, const float* __restrict__ W,
    const float* __restrict__ bias, float* __restrict__ C,
    int M, int N, int K, int do_sigmoid) {
  __shared__ __align__(16) bf16_t As[64][32 + LPAD];   // [m][k]
  __shared__ __align__(16) bf16_t Bs[64][32 + LPAD];   // [n][k]  (W transposed)
  const int tid  = threadIdx.x;
  const int w    = tid >> 5, lane = tid & 31, l15 = lane & 15;
  const int m0   = blockIdx.x * 64, n0 = blockIdx.y * 64;
  const int mt   = w & 3, npair = w >> 2;
  const int arow = tid >> 2, acol = (tid & 3) * 8;     // A tile loader mapping
  const int kr2  = (tid >> 4) * 2, ncl = (tid & 15) * 4; // W tile loader mapping
  const int ka   = (lane & 16) ? 16 : 0;               // A frag K sub-offset
  const int kb   = (lane & 16) ? 16 : 0;               // B frag K sub-offset
  v8f acc0 = {}; v8f acc1 = {};

  for (int k0 = 0; k0 < K; k0 += 32) {
    __syncthreads();
    // stage A tile: one async b128 global->LDS transfer per lane
    async_copy_b128(&As[arow][acol], &A[(size_t)(m0 + arow) * K + k0 + acol]);
    // stage W tile transposed, fp32 -> bf16, packed b32 LDS stores
    const float* wp0 = &W[(size_t)(k0 + kr2) * N + n0 + ncl];
    float4 r0 = *reinterpret_cast<const float4*>(wp0);
    float4 r1 = *reinterpret_cast<const float4*>(wp0 + N);
    union { unsigned u; bf16_t h[2]; } p0, p1, p2, p3;
    p0.h[0] = (bf16_t)r0.x; p0.h[1] = (bf16_t)r1.x;
    p1.h[0] = (bf16_t)r0.y; p1.h[1] = (bf16_t)r1.y;
    p2.h[0] = (bf16_t)r0.z; p2.h[1] = (bf16_t)r1.z;
    p3.h[0] = (bf16_t)r0.w; p3.h[1] = (bf16_t)r1.w;
    *reinterpret_cast<unsigned*>(&Bs[ncl + 0][kr2]) = p0.u;
    *reinterpret_cast<unsigned*>(&Bs[ncl + 1][kr2]) = p1.u;
    *reinterpret_cast<unsigned*>(&Bs[ncl + 2][kr2]) = p2.u;
    *reinterpret_cast<unsigned*>(&Bs[ncl + 3][kr2]) = p3.u;
    if (k0 + 32 < K)
      __builtin_prefetch(wp0 + (size_t)32 * N, 0, 0);
    wait_async0();
    __syncthreads();

    const int am = mt * 16 + l15;
    v16bf af = frag16(&As[am][ka], &As[am][ka + 8]);
    const int nA = npair * 32 + l15;
    v16bf bf0 = frag16(&Bs[nA][kb],      &Bs[nA][kb + 8]);
    v16bf bf1 = frag16(&Bs[nA + 16][kb], &Bs[nA + 16][kb + 8]);
    acc0 = __builtin_amdgcn_wmma_f32_16x16x32_bf16(false, af, false, bf0, (short)0, acc0, false, false);
    acc1 = __builtin_amdgcn_wmma_f32_16x16x32_bf16(false, af, false, bf1, (short)0, acc1, false, false);
  }

  const int rowb = m0 + mt * 16 + ((lane & 16) ? 8 : 0);
  const int c0   = n0 + npair * 32 + l15;
  const float b0v = bias[c0], b1v = bias[c0 + 16];
#pragma unroll
  for (int r = 0; r < 8; r++) {
    float v0 = acc0[r] + b0v;
    float v1 = acc1[r] + b1v;
    if (do_sigmoid) {
      v0 = 1.0f / (1.0f + __expf(-v0));
      v1 = 1.0f / (1.0f + __expf(-v1));
    }
    C[(size_t)(rowb + r) * N + c0]      = v0;
    C[(size_t)(rowb + r) * N + c0 + 16] = v1;
  }
}

// -------------------------------------- per-head L2 normalize (fp32->bf16)
__global__ __launch_bounds__(256) void normalize_heads_kernel(
    const float* __restrict__ in, bf16_t* __restrict__ out, int nseg) {
  int seg  = blockIdx.x * 8 + (threadIdx.x >> 5);
  int lane = threadIdx.x & 31;
  if (seg >= nseg) return;
  size_t base = (size_t)seg * DHEAD;
  float v0 = in[base + lane], v1 = in[base + 32 + lane];
  float ss = wave_sum(v0 * v0 + v1 * v1);
  float scale = 1.0f / fmaxf(sqrtf(ss), 1e-12f);
  out[base + lane]      = (bf16_t)(v0 * scale);
  out[base + 32 + lane] = (bf16_t)(v1 * scale);
}

// ---------------------------- v [B,S,D] fp32 -> vt [B,H,dh,S] bf16
__global__ __launch_bounds__(256) void transpose_v_kernel(
    const float* __restrict__ v, bf16_t* __restrict__ vt, int n) {
  int i = blockIdx.x * 256 + threadIdx.x;
  if (i >= n) return;
  int s = i % SEQ; int t = i / SEQ;
  int d = t % DHEAD; t /= DHEAD;
  int h = t % NHEADS; int b = t / NHEADS;
  vt[i] = (bf16_t)v[((size_t)(b * SEQ) + s) * DMODEL + h * DHEAD + d];
}

// ------------------------------------------- adaptive temperature -> 1/temp
__global__ __launch_bounds__(256) void temp_kernel(
    const float* __restrict__ x, const float* __restrict__ Wt,
    const float* __restrict__ bt, float* __restrict__ invtemp) {
  int b = blockIdx.x, tid = threadIdx.x;
  float part = 0.f;
  for (int d = tid; d < DMODEL; d += 256) {
    float s = 0.f;
    for (int i = 0; i < SEQ; i++) s += x[((size_t)b * SEQ + i) * DMODEL + d];
    part += (s * (1.0f / SEQ)) * Wt[d];
  }
  __shared__ float red[256];
  red[tid] = part; __syncthreads();
  for (int st = 128; st > 0; st >>= 1) {
    if (tid < st) red[tid] += red[tid + st];
    __syncthreads();
  }
  if (tid == 0) {
    float t = 1.0f / (1.0f + __expf(-(red[0] + bt[0]))) + 0.5f;
    invtemp[b] = 1.0f / t;
  }
}

// --------------------------------------------------------------- attention
// One block per 16-query tile of one (b,h). 8 waves.
__global__ __launch_bounds__(256) void attention_kernel(
    const bf16_t* __restrict__ qn, const bf16_t* __restrict__ kn,
    const bf16_t* __restrict__ vt, const float* __restrict__ invtemp,
    float* __restrict__ attn_out) {
  __shared__ __align__(16) bf16_t sc[16][SEQ + LPAD];   // ~64 KB score/weight tile
  __shared__ float outb[2][16][DHEAD];                  // 8 KB split-K partials
  const int tid = threadIdx.x, w = tid >> 5, lane = tid & 31, l15 = lane & 15;
  const int nq = SEQ / 16;
  const int qt = blockIdx.x % nq;
  const int h  = (blockIdx.x / nq) % NHEADS;
  const int b  = blockIdx.x / (nq * NHEADS);
  const int q0 = qt * 16;
  const float invT = invtemp[b];
  const int ka = (lane & 16) ? 16 : 0;   // A-frag K sub-offset
  const int kb = (lane & 16) ? 16 : 0;   // B-frag K sub-offset
  const int rb = (lane & 16) ? 8 : 0;    // C/D row sub-offset

  // q-tile A fragments (K = dh = 64 -> two k-steps), held in VGPRs
  const bf16_t* qbase = qn + ((size_t)(b * SEQ + q0 + l15)) * DMODEL + h * DHEAD;
  v16bf aq0 = frag16(qbase + ka,      qbase + ka + 8);
  v16bf aq1 = frag16(qbase + 32 + ka, qbase + 32 + ka + 8);

  // scores = (qn . kn^T) * invT  ->  LDS, bf16
  for (int it = 0; it < 16; it++) {
    int key = it * 128 + w * 16 + l15;
    const bf16_t* kbase = kn + ((size_t)(b * SEQ + key)) * DMODEL + h * DHEAD;
    v16bf b0 = frag16(kbase + kb,      kbase + kb + 8);
    v16bf b1 = frag16(kbase + 32 + kb, kbase + 32 + kb + 8);
    v8f acc = {};
    acc = __builtin_amdgcn_wmma_f32_16x16x32_bf16(false, aq0, false, b0, (short)0, acc, false, false);
    acc = __builtin_amdgcn_wmma_f32_16x16x32_bf16(false, aq1, false, b1, (short)0, acc, false, false);
#pragma unroll
    for (int r = 0; r < 8; r++) sc[rb + r][key] = (bf16_t)(acc[r] * invT);
  }
  __syncthreads();

  // top-k(512) per row via value bisection, then softmax over survivors.
  // All scans vectorized as v8bf (b128) LDS loads.
  for (int i = 0; i < 2; i++) {
    int r = w * 2 + i;
    float mx = -1e30f, mn = 1e30f;
    for (int j8 = lane; j8 < SEQ / 8; j8 += 32) {
      v8bf v = *reinterpret_cast<const v8bf*>(&sc[r][j8 * 8]);
#pragma unroll
      for (int e = 0; e < 8; e++) {
        float f = (float)v[e];
        mx = fmaxf(mx, f); mn = fminf(mn, f);
      }
    }
    mx = wave_max(mx); mn = -wave_max(-mn);
    float lo = mn, hi = mx;
    for (int itr = 0; itr < 22; itr++) {
      float mid = 0.5f * (lo + hi);
      int c = 0;
      for (int j8 = lane; j8 < SEQ / 8; j8 += 32) {
        v8bf v = *reinterpret_cast<const v8bf*>(&sc[r][j8 * 8]);
#pragma unroll
        for (int e = 0; e < 8; e++) c += ((float)v[e] >= mid) ? 1 : 0;
      }
      c = wave_isum(c);
      if (c >= TOPK) lo = mid; else hi = mid;
    }
    float thr = lo;
    float s = 0.f;
    for (int j8 = lane; j8 < SEQ / 8; j8 += 32) {
      v8bf v = *reinterpret_cast<const v8bf*>(&sc[r][j8 * 8]);
#pragma unroll
      for (int e = 0; e < 8; e++) {
        float f = (float)v[e];
        if (f >= thr) s += __expf(f - mx);
      }
    }
    s = wave_sum(s);
    float inv = 1.0f / s;
    for (int j8 = lane; j8 < SEQ / 8; j8 += 32) {
      v8bf v = *reinterpret_cast<const v8bf*>(&sc[r][j8 * 8]);
      v8bf o;
#pragma unroll
      for (int e = 0; e < 8; e++) {
        float f = (float)v[e];
        o[e] = (f >= thr) ? (bf16_t)(__expf(f - mx) * inv) : (bf16_t)0.0f;
      }
      *reinterpret_cast<v8bf*>(&sc[r][j8 * 8]) = o;
    }
  }
  __syncthreads();

  // out = attn @ v   (split K across two wave groups)
  {
    int nt = w & 3, half = w >> 2;
    const bf16_t* vbase =
        vt + ((size_t)((b * NHEADS + h) * DHEAD) + nt * 16 + l15) * SEQ;
    v8f acc = {};
    for (int kk = half * 1024; kk < half * 1024 + 1024; kk += 32) {
      v16bf a  = frag16(&sc[l15][kk + ka], &sc[l15][kk + ka + 8]);
      v16bf bb = frag16(vbase + kk + kb,   vbase + kk + kb + 8);
      acc = __builtin_amdgcn_wmma_f32_16x16x32_bf16(false, a, false, bb, (short)0, acc, false, false);
    }
#pragma unroll
    for (int r = 0; r < 8; r++) outb[half][rb + r][nt * 16 + l15] = acc[r];
  }
  __syncthreads();

  for (int idx = tid; idx < 16 * DHEAD; idx += 256) {
    int row = idx / DHEAD, col = idx % DHEAD;
    attn_out[((size_t)(b * SEQ + q0 + row)) * DMODEL + h * DHEAD + col] =
        outb[0][row][col] + outb[1][row][col];
  }
}

// ---------------------------------------------------- highway gate epilogue
// g already passed through sigmoid in the gate GEMM epilogue.
__global__ __launch_bounds__(256) void final_kernel(
    const float* __restrict__ x, const float* __restrict__ o,
    const float* __restrict__ g, float* __restrict__ out, int n) {
  int i = blockIdx.x * 256 + threadIdx.x;
  if (i < n) {
    float gg = g[i];
    out[i] = gg * o[i] + (1.0f - gg) * x[i];
  }
}

extern "C" void kernel_launch(void* const* d_in, const int* in_sizes, int n_in,
                              void* d_out, int out_size, void* d_ws, size_t ws_size,
                              hipStream_t stream) {
  (void)in_sizes; (void)n_in; (void)out_size; (void)ws_size;
  const float* x  = (const float*)d_in[0];
  const float* Wq = (const float*)d_in[1];
  const float* bq = (const float*)d_in[2];
  const float* Wk = (const float*)d_in[3];
  const float* bk = (const float*)d_in[4];
  const float* Wv = (const float*)d_in[5];
  const float* bv = (const float*)d_in[6];
  const float* Wo = (const float*)d_in[7];
  const float* bo = (const float*)d_in[8];
  const float* Wt = (const float*)d_in[9];
  const float* bt = (const float*)d_in[10];
  const float* Wg = (const float*)d_in[11];
  const float* bg = (const float*)d_in[12];
  float* out = (float*)d_out;

  char* ws = (char*)d_ws;
  const size_t SZ_BF = (size_t)TOT * 2;   // 8 MB
  const size_t SZ_F  = (size_t)TOT * 4;   // 16 MB
  bf16_t* xb   = (bf16_t*)(ws);
  float*  qf   = (float*) (ws + SZ_BF);
  float*  kf   = (float*) (ws + SZ_BF + 1 * SZ_F);
  float*  vf   = (float*) (ws + SZ_BF + 2 * SZ_F);
  bf16_t* qnb  = (bf16_t*)(ws + SZ_BF + 3 * SZ_F);
  bf16_t* knb  = (bf16_t*)(ws + 2 * SZ_BF + 3 * SZ_F);
  bf16_t* vtb  = (bf16_t*)(ws + 3 * SZ_BF + 3 * SZ_F);
  float*  itmp = (float*) (ws + 4 * SZ_BF + 3 * SZ_F);
  // dead-buffer reuse
  float*  aof = qf;    // attention output fp32 (q fp32 dead after normalize)
  bf16_t* aob = qnb;   // attention output bf16 (qn dead after attention)
  float*  of  = kf;    // O-projection output  (k fp32 dead after normalize)
  float*  gf  = vf;    // gate (sigmoided)     (v fp32 dead after transpose)

  dim3 g256((TOT + 255) / 256);
  dim3 gg(MROWS / 64, DMODEL / 64);

  f32_to_bf16_kernel<<<g256, 256, 0, stream>>>(x, xb, TOT);
  gemm_bf16_kernel<<<gg, 256, 0, stream>>>(xb, Wq, bq, qf, MROWS, DMODEL, DMODEL, 0);
  gemm_bf16_kernel<<<gg, 256, 0, stream>>>(xb, Wk, bk, kf, MROWS, DMODEL, DMODEL, 0);
  gemm_bf16_kernel<<<gg, 256, 0, stream>>>(xb, Wv, bv, vf, MROWS, DMODEL, DMODEL, 0);
  normalize_heads_kernel<<<(MROWS * NHEADS) / 8, 256, 0, stream>>>(qf, qnb, MROWS * NHEADS);
  normalize_heads_kernel<<<(MROWS * NHEADS) / 8, 256, 0, stream>>>(kf, knb, MROWS * NHEADS);
  transpose_v_kernel<<<g256, 256, 0, stream>>>(vf, vtb, TOT);
  temp_kernel<<<BATCH, 256, 0, stream>>>(x, Wt, bt, itmp);
  attention_kernel<<<BATCH * NHEADS * (SEQ / 16), 256, 0, stream>>>(qnb, knb, vtb, itmp, aof);
  f32_to_bf16_kernel<<<g256, 256, 0, stream>>>(aof, aob, TOT);
  gemm_bf16_kernel<<<gg, 256, 0, stream>>>(aob, Wo, bo, of, MROWS, DMODEL, DMODEL, 0);
  gemm_bf16_kernel<<<gg, 256, 0, stream>>>(xb, Wg, bg, gf, MROWS, DMODEL, DMODEL, 1);
  final_kernel<<<g256, 256, 0, stream>>>(x, of, gf, out, TOT);
}
